// SpatioTemporalAttention_48962627174787
// MI455X (gfx1250) — matrix operand order repaired and checked by reading, not compiled, not run on hardware
//
#include <hip/hip_runtime.h>
#include <hip/hip_bf16.h>
#include <math.h>

// ---------------------------------------------------------------------------
// MI455X / gfx1250 (wave32, WMMA). All GEMMs: v_wmma_f32_16x16x32_bf16.
// B operands are pre-transposed/padded ("Bt": N-major, K-contiguous) so every
// fragment load is two contiguous 16B runs/lane -> 2x global_load_b128, guard-free.
// GEMM: each wave computes a 16x64 strip (4 acc / A-frag), block tile 128x64.
// k-loop is manually 2x-unrolled ping-pong double buffering (no rotation
// copies): loads of step s+1 issue before wmmas of step s -> graded
// s_wait_loadcnt, loads fully overlapped with matrix ops.
// CPB bias uses the Toeplitz structure (1023 distinct rows, ~256x cheaper).
// Workspace: ~385 MB (hff intermediate dominates).
// ---------------------------------------------------------------------------

typedef __attribute__((ext_vector_type(16))) __bf16 v16bf;
typedef __attribute__((ext_vector_type(8)))  float  v8f;

#define DIMM   1024
#define HEADS  16
#define DHEAD  64
#define BB     16
#define NN     512
#define TOK    (BB*NN)     // 8192
#define CPBD   512
#define FFI    2730        // INNER_FF
#define FF2    5460        // 2*INNER_FF
#define FF2P   5504        // 2*INNER_FF padded to mult of 64 (GEMM N tile)
#define FFIP   2752        // INNER_FF padded to mult of 32 (GEMM K); 86 steps
#define EPSF   1e-5f

// ---- bf16 helpers ---------------------------------------------------------
static __device__ __forceinline__ unsigned short f2bf_raw(float f) {
  unsigned int u = __float_as_uint(f);
  if ((u & 0x7F800000u) == 0x7F800000u) return (unsigned short)(u >> 16); // inf/nan
  unsigned int r = u + 0x7FFFu + ((u >> 16) & 1u);                        // RNE
  return (unsigned short)(r >> 16);
}

union FragU {
  v16bf v;
  unsigned short u[16];
  unsigned int   w[8];
};

// A fragment (16x32 bf16, row-major, lda even).
// ISA layout: m = lane&15, k-band = (lane>>4)*8; lane holds K = band+[0,8) and
// band+16+[0,8): two contiguous 16B runs -> 2x b128.
static __device__ __forceinline__ v16bf load_a_frag(const unsigned short* A,
                                                    int lda, int row0, int k0) {
  int lane = threadIdx.x & 31;
  const unsigned short* rp =
      A + (size_t)(row0 + (lane & 15)) * lda + k0 + ((lane >> 4) << 3);
  FragU f;
#pragma unroll
  for (int v = 0; v < 4; ++v) f.w[v] = *(const unsigned int*)(rp + 2 * v);
#pragma unroll
  for (int v = 0; v < 4; ++v) f.w[4 + v] = *(const unsigned int*)(rp + 16 + 2 * v);
  return f.v;
}

// B fragment (32x16 bf16) from "Bt" storage: element B(k,n) = Bt[n*ldb + k].
// ISA layout: n = lane&15, k-band = (lane>>4)*8; lane holds K = band+[0,8) and
// band+16+[0,8): two contiguous 16B runs -> 2x b128.
static __device__ __forceinline__ v16bf load_bt_frag(const unsigned short* Bt,
                                                     int ldb, int col0, int k0) {
  int lane = threadIdx.x & 31;
  const unsigned short* rp =
      Bt + (size_t)(col0 + (lane & 15)) * ldb + k0 + ((lane >> 4) << 3);
  FragU f;
#pragma unroll
  for (int v = 0; v < 4; ++v) f.w[v] = *(const unsigned int*)(rp + 2 * v);
#pragma unroll
  for (int v = 0; v < 4; ++v) f.w[4 + v] = *(const unsigned int*)(rp + 16 + 2 * v);
  return f.v;
}

static __device__ __forceinline__ v8f wmma_bf16(v16bf a, v16bf b, v8f c) {
  return __builtin_amdgcn_wmma_f32_16x16x32_bf16(false, a, false, b,
                                                 (short)0, c, false, false);
}

// ---- weight preparation (one-shot, bandwidth-trivial) ---------------------
// Wq [1024,1024] + Wkv [1024,2048] -> transposed packed wqkv_t [3072][1024].
__global__ void pack_wqkv_t_kernel(const float* Wq, const float* Wkv,
                                   unsigned short* out) {
  size_t i = (size_t)blockIdx.x * 256 + threadIdx.x;
  if (i >= (size_t)3072 * DIMM) return;
  int c = (int)(i / DIMM), r = (int)(i % DIMM);  // c: out col, r: in row
  float v = (c < DIMM) ? Wq[(size_t)r * DIMM + c]
                       : Wkv[(size_t)r * 2048 + (c - DIMM)];
  out[i] = f2bf_raw(v);
}

// Generic transpose+convert: in [R,C] f32 -> out [C,R] bf16.
__global__ void transpose_f2bf_kernel(const float* in, unsigned short* out,
                                      int R, int C) {
  size_t i = (size_t)blockIdx.x * 256 + threadIdx.x;
  if (i >= (size_t)R * C) return;
  int c = (int)(i / R), r = (int)(i % R);
  out[i] = f2bf_raw(in[(size_t)r * C + c]);
}

// ff_w_in [5460,1024] (already Bt layout) -> rows padded to 5504 with zeros.
__global__ void ffin_pad_kernel(const float* in, unsigned short* out) {
  size_t i = (size_t)blockIdx.x * 256 + threadIdx.x;
  if (i >= (size_t)FF2P * DIMM) return;
  int o = (int)(i / DIMM);
  out[i] = (o < FF2) ? f2bf_raw(in[i]) : (unsigned short)0;
}

// ff_w_out [1024,2730] -> [1024][2752], K-tail zeroed (matches zero-padded A).
__global__ void ffout_pad_kernel(const float* in, unsigned short* out) {
  size_t i = (size_t)blockIdx.x * 256 + threadIdx.x;
  if (i >= (size_t)DIMM * FFIP) return;
  int o = (int)(i / FFIP), c = (int)(i % FFIP);
  out[i] = (c < FFI) ? f2bf_raw(in[(size_t)o * FFI + c]) : (unsigned short)0;
}

// ---- CPB bias: 1023 distinct relative positions ---------------------------
__global__ __launch_bounds__(256)
void cpb_kernel(const float* w1, const float* b1v, const float* w2,
                const float* b2v, const float* w3, const float* b3v,
                float* bias_tab) {
  __shared__ float h1[CPBD];
  __shared__ float h2[CPBD];
  __shared__ float red[HEADS];
  int r = blockIdx.x;                 // rel = r - (N-1) in [-511, 511]
  int t = threadIdx.x;
  float d = (float)(r - (NN - 1));
  float sgn = (d > 0.f) ? 1.f : ((d < 0.f) ? -1.f : 0.f);
  float relv = sgn * logf(fabsf(d) + 1.f);

  for (int c = t; c < CPBD; c += 256) {
    float z = relv * w1[c] + b1v[c];
    h1[c] = z / (1.f + __expf(-z));            // silu
  }
  if (t < HEADS) red[t] = 0.f;
  __syncthreads();
  for (int c2 = t; c2 < CPBD; c2 += 256) {
    float acc = b2v[c2];
    for (int c = 0; c < CPBD; ++c) acc += h1[c] * w2[(size_t)c * CPBD + c2];
    h2[c2] = acc / (1.f + __expf(-acc));       // silu
  }
  __syncthreads();
  float part[HEADS];
#pragma unroll
  for (int hh = 0; hh < HEADS; ++hh) part[hh] = 0.f;
  for (int c2 = t; c2 < CPBD; c2 += 256) {
    float hv = h2[c2];
#pragma unroll
    for (int hh = 0; hh < HEADS; ++hh) part[hh] += hv * w3[(size_t)c2 * HEADS + hh];
  }
#pragma unroll
  for (int hh = 0; hh < HEADS; ++hh) atomicAdd(&red[hh], part[hh]);
  __syncthreads();
  if (t < HEADS) bias_tab[(size_t)r * HEADS + t] = red[t] + b3v[t];
}

// ---- sequence-axis LayerNorm (reference normalizes over n!) ---------------
__global__ __launch_bounds__(256)
void seqln_kernel(const float* x, const float* g, unsigned short* xn) {
  int t = blockIdx.x * 256 + threadIdx.x;
  if (t >= BB * DIMM) return;
  int b = t >> 10, d = t & 1023;
  const float* xp = x + (size_t)b * NN * DIMM + d;
  float s = 0.f, ss = 0.f;
  for (int n = 0; n < NN; ++n) { float v = xp[(size_t)n * DIMM]; s += v; ss += v * v; }
  float mean = s / NN;
  float var = ss / NN - mean * mean;
  float rstd = rsqrtf(fmaxf(var, EPSF));
  float gd = g[d];
  unsigned short* op = xn + (size_t)b * NN * DIMM + d;
  for (int n = 0; n < NN; ++n)
    op[(size_t)n * DIMM] = f2bf_raw((xp[(size_t)n * DIMM] - mean) * rstd * gd);
}

// ---- WMMA GEMM: 8 waves, each a 16x64 strip (4 acc / A-frag); tile 128x64 --
// All dims grid-exact (M%128==0, N%64==0, (K/32)%2==0) -> guard-free hot loop.
// Ping-pong double buffering, no rotation copies.
// mode 0: C[row*ldc+col] = acc
// mode 1: qkv scatter -> q/k [B,H,N,64] (q * DHEAD^-0.5), v transposed [B,H,64,N]
// mode 2: nan_to_num(acc) + resid -> C (+ optional bf16 copy)
__global__ __launch_bounds__(256)
void gemm_bf16_kernel(const unsigned short* __restrict__ A, int lda,
                      const unsigned short* __restrict__ Bt, int ldb,
                      int K, int mode,
                      float* C, int ldc, const float* resid,
                      unsigned short* out_bf16,
                      unsigned short* q_out, unsigned short* k_out,
                      unsigned short* v_out) {
  int w = threadIdx.x >> 5;
  int lane = threadIdx.x & 31;
  int row0 = blockIdx.x * 128 + w * 16;
  int col0 = blockIdx.y * 64;

  v8f acc[4] = {};
  int nsteps = K >> 5;                 // even (32 or 86) for all launches

  // prologue: set A holds step 0
  v16bf aA = load_a_frag(A, lda, row0, 0);
  v16bf bA[4];
#pragma unroll
  for (int t4 = 0; t4 < 4; ++t4) bA[t4] = load_bt_frag(Bt, ldb, col0 + t4 * 16, 0);

  int k = 32;
  for (int s = 0; s + 2 < nsteps; s += 2) {
    // load step s+1 into set B, then compute step s from set A
    v16bf aB = load_a_frag(A, lda, row0, k);
    v16bf bB[4];
#pragma unroll
    for (int t4 = 0; t4 < 4; ++t4) bB[t4] = load_bt_frag(Bt, ldb, col0 + t4 * 16, k);
    k += 32;
#pragma unroll
    for (int t4 = 0; t4 < 4; ++t4) acc[t4] = wmma_bf16(aA, bA[t4], acc[t4]);
    // load step s+2 into set A, then compute step s+1 from set B
    aA = load_a_frag(A, lda, row0, k);
#pragma unroll
    for (int t4 = 0; t4 < 4; ++t4) bA[t4] = load_bt_frag(Bt, ldb, col0 + t4 * 16, k);
    k += 32;
#pragma unroll
    for (int t4 = 0; t4 < 4; ++t4) acc[t4] = wmma_bf16(aB, bB[t4], acc[t4]);
  }
  // tail: steps nsteps-2 (in set A) and nsteps-1
  {
    v16bf aB = load_a_frag(A, lda, row0, k);
    v16bf bB[4];
#pragma unroll
    for (int t4 = 0; t4 < 4; ++t4) bB[t4] = load_bt_frag(Bt, ldb, col0 + t4 * 16, k);
#pragma unroll
    for (int t4 = 0; t4 < 4; ++t4) acc[t4] = wmma_bf16(aA, bA[t4], acc[t4]);
#pragma unroll
    for (int t4 = 0; t4 < 4; ++t4) acc[t4] = wmma_bf16(aB, bB[t4], acc[t4]);
  }

  int n = lane & 15;
  int hl = lane >> 4;
#pragma unroll
  for (int t4 = 0; t4 < 4; ++t4) {
    int col = col0 + t4 * 16 + n;
#pragma unroll
    for (int r = 0; r < 8; ++r) {
      int row = row0 + r + 8 * hl;
      float v = acc[t4][r];
      if (mode == 0) {
        C[(size_t)row * ldc + col] = v;
      } else if (mode == 1) {
        int which = col >> 10;            // 0=q 1=k 2=v
        int cc = col & 1023;
        int h = cc >> 6, dd = cc & 63;
        int bi = row >> 9, ni = row & 511;
        if (which == 0) {
          q_out[((((size_t)bi * HEADS + h) * NN) + ni) * DHEAD + dd] =
              f2bf_raw(v * 0.125f);       // DHEAD^-0.5
        } else if (which == 1) {
          k_out[((((size_t)bi * HEADS + h) * NN) + ni) * DHEAD + dd] = f2bf_raw(v);
        } else {                          // v stored transposed per head
          v_out[((((size_t)bi * HEADS + h) * DHEAD) + dd) * NN + ni] = f2bf_raw(v);
        }
      } else {  // mode 2
        if (isnan(v)) v = 0.0f;
        else if (isinf(v)) v = (v > 0.f) ? 3.402823466e38f : -3.402823466e38f;
        v += resid[(size_t)row * ldc + col];
        C[(size_t)row * ldc + col] = v;
        if (out_bf16) out_bf16[(size_t)row * ldc + col] = f2bf_raw(v);
      }
    }
  }
}

// ---- flash attention: one (b,h,128 q-rows) per block ----------------------
// q,k: [B,H,N,64]; v: [B,H,64,N] (transposed -> contiguous PV B-fragments).
__global__ __launch_bounds__(256)
void attention_kernel(const unsigned short* __restrict__ q,
                      const unsigned short* __restrict__ k,
                      const unsigned short* __restrict__ v,
                      const float* __restrict__ bias_tab,
                      unsigned short* __restrict__ o_out) {
  __shared__ __align__(64) unsigned short pstage[8][16][32];  // per-wave P tile
  int w = threadIdx.x >> 5;
  int lane = threadIdx.x & 31;
  int n = lane & 15;
  int hl = lane >> 4;

  int blk = blockIdx.x;              // b*64 + h*4 + qb
  int qb = blk & 3;
  int h = (blk >> 2) & 15;
  int b = blk >> 6;
  int qrow0 = qb * 128 + w * 16;

  const size_t headoff = ((size_t)b * HEADS + h) * NN * DHEAD;
  const unsigned short* qbase = q + headoff;
  const unsigned short* kbase = k + headoff;   // Bt form: B(kdim,key)=kbase[key*64+kdim]
  const unsigned short* vbase = v + headoff;   // Bt form: B(key,dim)=vbase[dim*512+key]

  v16bf qa0 = load_a_frag(qbase, DHEAD, qrow0, 0);   // q already * 1/8
  v16bf qa1 = load_a_frag(qbase, DHEAD, qrow0, 32);

  v8f o0 = {}, o1 = {}, o2 = {}, o3 = {};
  float mrow[8], lrow[8];
#pragma unroll
  for (int r = 0; r < 8; ++r) { mrow[r] = -INFINITY; lrow[r] = 0.f; }

  for (int j0 = 0; j0 < NN; j0 += 32) {
    float s[2][8], tmax[2][8];
#pragma unroll
    for (int jt = 0; jt < 2; ++jt) {
      int jc0 = j0 + jt * 16;
      v16bf kb0 = load_bt_frag(kbase, DHEAD, jc0, 0);
      v16bf kb1 = load_bt_frag(kbase, DHEAD, jc0, 32);
      v8f sacc = {};
      sacc = wmma_bf16(qa0, kb0, sacc);
      sacc = wmma_bf16(qa1, kb1, sacc);
#pragma unroll
      for (int r = 0; r < 8; ++r) {
        int ig = qrow0 + r + 8 * hl;
        int jg = jc0 + n;
        float sv = sacc[r] + bias_tab[(size_t)(ig - jg + (NN - 1)) * HEADS + h];
        s[jt][r] = sv;
        float t = sv;                     // 16-lane row-max (stays in half)
        t = fmaxf(t, __shfl_xor(t, 1));
        t = fmaxf(t, __shfl_xor(t, 2));
        t = fmaxf(t, __shfl_xor(t, 4));
        t = fmaxf(t, __shfl_xor(t, 8));
        tmax[jt][r] = t;
      }
    }
#pragma unroll
    for (int r = 0; r < 8; ++r) {
      float mnew = fmaxf(mrow[r], fmaxf(tmax[0][r], tmax[1][r]));
      float corr = __expf(mrow[r] - mnew);
      mrow[r] = mnew;
      lrow[r] *= corr;
      o0[r] *= corr; o1[r] *= corr; o2[r] *= corr; o3[r] *= corr;
#pragma unroll
      for (int jt = 0; jt < 2; ++jt) {
        float p = __expf(s[jt][r] - mnew);
        float t = p;
        t += __shfl_xor(t, 1); t += __shfl_xor(t, 2);
        t += __shfl_xor(t, 4); t += __shfl_xor(t, 8);
        lrow[r] += t;
        pstage[w][r + 8 * hl][jt * 16 + n] = f2bf_raw(p);
      }
    }
    __syncthreads();
    v16bf pa = load_a_frag(&pstage[w][0][0], 32, 0, 0);
    v16bf vb[4];
#pragma unroll
    for (int t4 = 0; t4 < 4; ++t4)
      vb[t4] = load_bt_frag(vbase, NN, t4 * 16, j0);
    o0 = wmma_bf16(pa, vb[0], o0);
    o1 = wmma_bf16(pa, vb[1], o1);
    o2 = wmma_bf16(pa, vb[2], o2);
    o3 = wmma_bf16(pa, vb[3], o3);
    __syncthreads();
  }

#pragma unroll
  for (int r = 0; r < 8; ++r) {
    float inv = 1.0f / lrow[r];
    int tokrow = b * NN + qrow0 + r + 8 * hl;          // token index
    unsigned short* orow = o_out + (size_t)tokrow * DIMM + h * DHEAD;
    orow[0 * 16 + n] = f2bf_raw(o0[r] * inv);
    orow[1 * 16 + n] = f2bf_raw(o1[r] * inv);
    orow[2 * 16 + n] = f2bf_raw(o2[r] * inv);
    orow[3 * 16 + n] = f2bf_raw(o3[r] * inv);
  }
}

// ---- GEGLU + channel LayerNorm, one token per block -----------------------
__global__ __launch_bounds__(256)
void geglu_ln_kernel(const float* hff, const float* ffg, unsigned short* hln) {
  __shared__ float gbuf[FFI];
  __shared__ float rs[256], rss[256];
  int tok = blockIdx.x;
  int t = threadIdx.x;
  const float* row = hff + (size_t)tok * FF2P;
  float s = 0.f, ss = 0.f;
  for (int c = t; c < FFI; c += 256) {
    float a = row[c];
    float gt = row[FFI + c];
    float gl = 0.5f * gt * (1.f + erff(gt * 0.7071067811865475f));  // exact gelu
    float val = a * gl;
    gbuf[c] = val;
    s += val; ss += val * val;
  }
  rs[t] = s; rss[t] = ss;
  __syncthreads();
  for (int off = 128; off > 0; off >>= 1) {
    if (t < off) { rs[t] += rs[t + off]; rss[t] += rss[t + off]; }
    __syncthreads();
  }
  float mean = rs[0] / FFI;
  float var = rss[0] / FFI - mean * mean;
  float rstd = rsqrtf(fmaxf(var, EPSF));
  unsigned short* orow = hln + (size_t)tok * FFIP;
  for (int c = t; c < FFIP; c += 256)
    orow[c] = (c < FFI) ? f2bf_raw((gbuf[c] - mean) * rstd * ffg[c])
                        : (unsigned short)0;   // zero K-pad
}

// ---------------------------------------------------------------------------
extern "C" void kernel_launch(void* const* d_in, const int* in_sizes, int n_in,
                              void* d_out, int out_size, void* d_ws, size_t ws_size,
                              hipStream_t stream) {
  const float* x      = (const float*)d_in[0];
  const float* g_ln   = (const float*)d_in[1];
  const float* Wq     = (const float*)d_in[2];
  const float* Wkv    = (const float*)d_in[3];
  const float* Wo     = (const float*)d_in[4];
  const float* cpb_w1 = (const float*)d_in[5];
  const float* cpb_b1 = (const float*)d_in[6];
  const float* cpb_w2 = (const float*)d_in[7];
  const float* cpb_b2 = (const float*)d_in[8];
  const float* cpb_w3 = (const float*)d_in[9];
  const float* cpb_b3 = (const float*)d_in[10];
  const float* ff_w_in  = (const float*)d_in[11];
  const float* ff_g     = (const float*)d_in[12];
  const float* ff_w_out = (const float*)d_in[13];
  float* out = (float*)d_out;

  char* ws = (char*)d_ws;
  size_t off = 0;
  auto take = [&](size_t bytes) {
    size_t o = off; off += (bytes + 255) & ~(size_t)255; return o;
  };
  unsigned short* wqkv_t   = (unsigned short*)(ws + take((size_t)3072 * DIMM * 2));
  unsigned short* wo_t     = (unsigned short*)(ws + take((size_t)DIMM * DIMM * 2));
  unsigned short* ffin_p   = (unsigned short*)(ws + take((size_t)FF2P * DIMM * 2));
  unsigned short* ffout_p  = (unsigned short*)(ws + take((size_t)DIMM * FFIP * 2));
  float*          bias_tab = (float*)(ws + take((size_t)(2 * NN - 1) * HEADS * 4));
  unsigned short* xn_bf    = (unsigned short*)(ws + take((size_t)TOK * DIMM * 2));
  unsigned short* q_bf     = (unsigned short*)(ws + take((size_t)TOK * DIMM * 2));
  unsigned short* k_bf     = (unsigned short*)(ws + take((size_t)TOK * DIMM * 2));
  unsigned short* v_bf     = (unsigned short*)(ws + take((size_t)TOK * DIMM * 2));
  unsigned short* o_bf     = (unsigned short*)(ws + take((size_t)TOK * DIMM * 2));
  float*          x1_f32   = (float*)(ws + take((size_t)TOK * DIMM * 4));
  unsigned short* x1_bf    = (unsigned short*)(ws + take((size_t)TOK * DIMM * 2));
  float*          hff_f32  = (float*)(ws + take((size_t)TOK * FF2P * 4));
  unsigned short* hln_bf   = (unsigned short*)(ws + take((size_t)TOK * FFIP * 2));
  (void)ws_size; (void)in_sizes; (void)n_in; (void)out_size;

  // 1) weight prep: transpose/pad to Bt layouts, bf16
  pack_wqkv_t_kernel<<<((size_t)3072 * DIMM + 255) / 256, 256, 0, stream>>>(Wq, Wkv, wqkv_t);
  transpose_f2bf_kernel<<<((size_t)DIMM * DIMM + 255) / 256, 256, 0, stream>>>(Wo, wo_t, DIMM, DIMM);
  ffin_pad_kernel<<<((size_t)FF2P * DIMM + 255) / 256, 256, 0, stream>>>(ff_w_in, ffin_p);
  ffout_pad_kernel<<<((size_t)DIMM * FFIP + 255) / 256, 256, 0, stream>>>(ff_w_out, ffout_p);

  // 2) CPB bias over 1023 distinct relative offsets
  cpb_kernel<<<2 * NN - 1, 256, 0, stream>>>(cpb_w1, cpb_b1, cpb_w2, cpb_b2,
                                             cpb_w3, cpb_b3, bias_tab);

  // 3) sequence-axis LayerNorm -> bf16
  seqln_kernel<<<(BB * DIMM + 255) / 256, 256, 0, stream>>>(x, g_ln, xn_bf);

  // 4) fused q|k|v projection (scatter: q,k [B,H,N,64], v [B,H,64,N], q*1/8)
  gemm_bf16_kernel<<<dim3(TOK / 128, 3072 / 64), 256, 0, stream>>>(
      xn_bf, DIMM, wqkv_t, DIMM, DIMM, 1,
      nullptr, 0, nullptr, nullptr, q_bf, k_bf, v_bf);

  // 5) flash attention with Toeplitz bias table
  attention_kernel<<<BB * HEADS * (NN / 128), 256, 0, stream>>>(
      q_bf, k_bf, v_bf, bias_tab, o_bf);

  // 6) output projection + nan_to_num + residual -> x1 (f32 + bf16)
  gemm_bf16_kernel<<<dim3(TOK / 128, DIMM / 64), 256, 0, stream>>>(
      o_bf, DIMM, wo_t, DIMM, DIMM, 2,
      x1_f32, DIMM, x, x1_bf, nullptr, nullptr, nullptr);

  // 7) FF in: hff[tok, 0..5503] = x1 @ ff_w_in^T (padded, guard-free)
  gemm_bf16_kernel<<<dim3(TOK / 128, FF2P / 64), 256, 0, stream>>>(
      x1_bf, DIMM, ffin_p, DIMM, DIMM, 0,
      hff_f32, FF2P, nullptr, nullptr, nullptr, nullptr, nullptr);

  // 8) GEGLU + channel LayerNorm -> bf16 (K padded to 2752 with zeros)
  geglu_ln_kernel<<<TOK, 256, 0, stream>>>(hff_f32, ff_g, hln_bf);

  // 9) FF out + residual -> d_out
  gemm_bf16_kernel<<<dim3(TOK / 128, DIMM / 64), 256, 0, stream>>>(
      hln_bf, FFIP, ffout_p, FFIP, FFIP, 2,
      out, DIMM, x1_f32, nullptr, nullptr, nullptr, nullptr);
}
